// MultiHeadSelfAttention_10179072492053
// MI455X (gfx1250) — matrix operand order repaired and checked by reading, not compiled, not run on hardware
//
#include <hip/hip_runtime.h>
#include <stdint.h>

typedef __bf16 bf16;
typedef __attribute__((ext_vector_type(16))) __bf16 v16bf;
typedef __attribute__((ext_vector_type(8)))  __bf16 v8bf;
typedef __attribute__((ext_vector_type(8)))  float  v8f;

#define EMB   768
#define HEADS 12
#define DH    64
#define BATCH 8
#define SEQ   1024
#define MTOT  (BATCH * SEQ)   // 8192

// ---------- bf16 helpers (round-to-nearest-even) ----------
__device__ __forceinline__ uint16_t bfbits(float f) {
  union { float f; uint32_t u; } x; x.f = f;
  uint32_t r = x.u + 0x7fffu + ((x.u >> 16) & 1u);
  return (uint16_t)(r >> 16);
}
__device__ __forceinline__ bf16 f2bf(float f) {
  uint16_t h = bfbits(f);
  return __builtin_bit_cast(bf16, h);
}
__device__ __forceinline__ uint32_t pk2(float a, float b) {
  return (uint32_t)bfbits(a) | ((uint32_t)bfbits(b) << 16);
}
__device__ __forceinline__ v8f zero8() {
  v8f v = {0.f,0.f,0.f,0.f,0.f,0.f,0.f,0.f}; return v;
}

// ---------- CDNA5 async global -> LDS copy (ASYNCcnt-tracked) ----------
// GV mode: per-lane 64-bit global address, LDS byte address in a VGPR.
// Generic shared pointer's low 32 bits == LDS offset (aperture rule).
__device__ __forceinline__ void async_b128(bf16* lds_dst, const bf16* gsrc) {
  uint32_t l = (uint32_t)(uintptr_t)lds_dst;
  asm volatile("global_load_async_to_lds_b128 %0, %1, off"
               :: "v"(l), "v"(gsrc) : "memory");
}
#define WAIT_ASYNC(imm) asm volatile("s_wait_asynccnt " imm ::: "memory")

// ---------- WMMA fragment loaders (ISA 7.12.2 layouts, wave32) ----------
__device__ __forceinline__ v16bf load_a_frag(const bf16* base, int stride) {
  int lane = threadIdx.x & 31;
  int row  = lane & 15;
  int kb0  = (lane >> 4) * 8;
  const bf16* p = base + row * stride;
  union { v16bf v; v8bf h[2]; } u;
  u.h[0] = *(const v8bf*)(p + kb0);
  u.h[1] = *(const v8bf*)(p + kb0 + 16);
  return u.v;
}
__device__ __forceinline__ v16bf load_b_frag(const bf16* base, int stride) {
  int lane = threadIdx.x & 31;
  int col  = lane & 15;
  int kb   = (lane >> 4) * 8;
  const bf16* p = base + col * stride + kb;
  union { v16bf v; v8bf h[2]; } u;
  u.h[0] = *(const v8bf*)(p);
  u.h[1] = *(const v8bf*)(p + 16);
  return u.v;
}
__device__ __forceinline__ v8f wmma_bf16(v16bf a, v16bf b, v8f c) {
  return __builtin_amdgcn_wmma_f32_16x16x32_bf16(false, a, false, b, (short)0, c,
                                                 false, false);
}

// ============================================================
// Kernel 0: f32 -> bf16 elementwise (8 elems/thread, exact sizes)
// ============================================================
__global__ __launch_bounds__(256) void k_cvt(const float* __restrict__ s,
                                             bf16* __restrict__ d) {
  size_t i = ((size_t)blockIdx.x * 256 + threadIdx.x) * 8;
  const float4* p = (const float4*)(s + i);
  float4 f0 = p[0], f1 = p[1];
  uint4 o;
  o.x = pk2(f0.x, f0.y); o.y = pk2(f0.z, f0.w);
  o.z = pk2(f1.x, f1.y); o.w = pk2(f1.z, f1.w);
  *(uint4*)(d + i) = o;
}

// ============================================================
// Kernel 1: Q/K/V projection.  C[8192,768] = zb @ W^T  (bf16)
// grid (768/64, 8192/128, 3), block 256 (8 waves), async double-buffered
// ============================================================
#define PM 128
#define PN 64
#define PKS 32
#define SAS 40
#define SBS 40

__global__ __launch_bounds__(256) void k_proj(
    const bf16* __restrict__ zb,
    const bf16* __restrict__ wqb, const bf16* __restrict__ wkb,
    const bf16* __restrict__ wvb,
    bf16* __restrict__ Q, bf16* __restrict__ Kd, bf16* __restrict__ Vt)
{
  __shared__ bf16 sA[2][PM * SAS];
  __shared__ bf16 sB[2][PN * SBS];

  const int t    = threadIdx.x;
  const int lane = t & 31;
  const int wid  = t >> 5;
  const int hi   = lane >> 4;

  const int j0   = blockIdx.x * PN;
  const int m0   = blockIdx.y * PM;
  const int mode = blockIdx.z;
  const bf16* W  = (mode == 0) ? wqb : (mode == 1) ? wkb : wvb;

  const int wm = (wid & 3) * 32;
  const int wn = (wid >> 2) * 32;

  v8f acc[2][2];
#pragma unroll
  for (int i = 0; i < 2; i++)
#pragma unroll
    for (int j = 0; j < 2; j++) acc[i][j] = zero8();

  // 3 async b128 per thread (per wave) per K-step
  auto issue = [&](int buf, int k0) {
#pragma unroll
    for (int it = 0; it < 2; it++) {
      int idx = t + it * 256;                 // A: 128x32 bf16 = 512 x b128
      int row = idx >> 2, c = (idx & 3) * 8;
      async_b128(&sA[buf][row * SAS + c],
                 zb + (size_t)(m0 + row) * EMB + k0 + c);
    }
    {
      int row = t >> 2, c = (t & 3) * 8;      // B: 64x32 bf16 = 256 x b128
      async_b128(&sB[buf][row * SBS + c],
                 W + (size_t)(j0 + row) * EMB + k0 + c);
    }
  };

  issue(0, 0);
  int cur = 0;
  for (int k0 = 0; k0 < EMB; k0 += PKS) {
    if (k0 + PKS < EMB) { issue(cur ^ 1, k0 + PKS); WAIT_ASYNC("0x3"); }
    else                {                           WAIT_ASYNC("0x0"); }
    __syncthreads();
    v16bf a0 = load_a_frag(&sA[cur][(wm + 0)  * SAS], SAS);
    v16bf a1 = load_a_frag(&sA[cur][(wm + 16) * SAS], SAS);
    v16bf b0 = load_b_frag(&sB[cur][(wn + 0)  * SBS], SBS);
    v16bf b1 = load_b_frag(&sB[cur][(wn + 16) * SBS], SBS);
    acc[0][0] = wmma_bf16(a0, b0, acc[0][0]);
    acc[0][1] = wmma_bf16(a0, b1, acc[0][1]);
    acc[1][0] = wmma_bf16(a1, b0, acc[1][0]);
    acc[1][1] = wmma_bf16(a1, b1, acc[1][1]);
    __syncthreads();
    cur ^= 1;
  }

#pragma unroll
  for (int sm = 0; sm < 2; sm++)
#pragma unroll
    for (int sn = 0; sn < 2; sn++)
#pragma unroll
      for (int r = 0; r < 8; r++) {
        int gm = m0 + wm + sm * 16 + r + hi * 8;
        int gj = j0 + wn + sn * 16 + (lane & 15);
        int b = gm >> 10, n = gm & 1023;
        int h = gj >> 6,  d = gj & 63;
        bf16 v = f2bf(acc[sm][sn][r]);
        if (mode == 2)
          Vt[((size_t)((b * HEADS + h) * DH + d)) * SEQ + n] = v;     // V^T
        else {
          bf16* dst = (mode == 0) ? Q : Kd;
          dst[((size_t)((b * HEADS + h) * SEQ + n)) * DH + d] = v;
        }
      }
}

// ============================================================
// Kernel 2: flash attention.  grid (1024/128, 96), block 256.
// Async double-buffered key/value chunks of 64.
// ============================================================
#define AKS 72

__global__ __launch_bounds__(256) void k_attn(
    const bf16* __restrict__ Q, const bf16* __restrict__ Kd,
    const bf16* __restrict__ Vt, bf16* __restrict__ Y)
{
  __shared__ bf16 sK[2][64 * AKS];     // [key][dh]
  __shared__ bf16 sV[2][64 * AKS];     // [dh][key]
  __shared__ bf16 sP[8 * 16 * AKS];    // per-wave P staging

  const int t    = threadIdx.x;
  const int lane = t & 31;
  const int w    = t >> 5;
  const int hi   = lane >> 4;
  const int col  = lane & 15;

  const int bh = blockIdx.y;
  const int b  = bh / HEADS, h = bh % HEADS;
  const int q0 = blockIdx.x * 128;

  const size_t headK = (size_t)bh * SEQ * DH;
  const size_t headV = (size_t)bh * DH * SEQ;

  const bf16* qbase = Q + headK + (size_t)(q0 + w * 16) * DH;
  v16bf qf[2];
  qf[0] = load_a_frag(qbase + 0,  DH);
  qf[1] = load_a_frag(qbase + 32, DH);

  v8f Oacc[4];
#pragma unroll
  for (int d = 0; d < 4; d++) Oacc[d] = zero8();
  float m_run[8], l_run[8];
#pragma unroll
  for (int r = 0; r < 8; r++) { m_run[r] = -3.0e38f; l_run[r] = 0.0f; }

  bf16* sPw = &sP[w * 16 * AKS];

  // 4 async b128 per thread (per wave) per chunk
  auto issue = [&](int buf, int n0) {
#pragma unroll
    for (int it = 0; it < 2; it++) {
      int idx = t + it * 256;
      int row = idx >> 3, c = (idx & 7) * 8;
      async_b128(&sK[buf][row * AKS + c],
                 Kd + headK + (size_t)(n0 + row) * DH + c);
      async_b128(&sV[buf][row * AKS + c],
                 Vt + headV + (size_t)row * SEQ + n0 + c);
    }
  };

  issue(0, 0);
  int cur = 0;
  for (int n0 = 0; n0 < SEQ; n0 += 64) {
    if (n0 + 64 < SEQ) { issue(cur ^ 1, n0 + 64); WAIT_ASYNC("0x4"); }
    else               {                          WAIT_ASYNC("0x0"); }
    __syncthreads();

    // S = (Q K^T) / sqrt(64)
    v8f S[4];
#pragma unroll
    for (int ns = 0; ns < 4; ns++) {
      S[ns] = zero8();
#pragma unroll
      for (int c = 0; c < 2; c++)
        S[ns] = wmma_bf16(qf[c],
                          load_b_frag(&sK[cur][(ns * 16) * AKS + c * 32], AKS),
                          S[ns]);
      S[ns] = S[ns] * 0.125f;
    }

    // online softmax; C layout: VGPR r <-> row r (+8 for upper half-wave)
#pragma unroll
    for (int r = 0; r < 8; r++) {
      float mx = fmaxf(fmaxf(S[0][r], S[1][r]), fmaxf(S[2][r], S[3][r]));
      mx = fmaxf(mx, __shfl_xor(mx, 1, 32));
      mx = fmaxf(mx, __shfl_xor(mx, 2, 32));
      mx = fmaxf(mx, __shfl_xor(mx, 4, 32));
      mx = fmaxf(mx, __shfl_xor(mx, 8, 32));
      float mnew = fmaxf(m_run[r], mx);
      float sc   = __expf(m_run[r] - mnew);
      float rs   = 0.0f;
#pragma unroll
      for (int ns = 0; ns < 4; ns++) {
        float p = __expf(S[ns][r] - mnew);
        rs += p;
        sPw[(r + hi * 8) * AKS + ns * 16 + col] = f2bf(p);
      }
      rs += __shfl_xor(rs, 1, 32);
      rs += __shfl_xor(rs, 2, 32);
      rs += __shfl_xor(rs, 4, 32);
      rs += __shfl_xor(rs, 8, 32);
      l_run[r] = l_run[r] * sc + rs;
      m_run[r] = mnew;
#pragma unroll
      for (int d = 0; d < 4; d++) Oacc[d][r] = Oacc[d][r] * sc;
    }
    asm volatile("s_wait_dscnt 0x0" ::: "memory");  // P stores -> P loads (wave-local)

    // O += P V
#pragma unroll
    for (int kc = 0; kc < 2; kc++) {
      v16bf pa = load_a_frag(&sPw[kc * 32], AKS);
#pragma unroll
      for (int d = 0; d < 4; d++)
        Oacc[d] = wmma_bf16(pa,
                            load_b_frag(&sV[cur][(d * 16) * AKS + kc * 32], AKS),
                            Oacc[d]);
    }
    __syncthreads();
    cur ^= 1;
  }

#pragma unroll
  for (int r = 0; r < 8; r++) {
    float inv = 1.0f / l_run[r];
    int qrow  = q0 + w * 16 + r + hi * 8;
#pragma unroll
    for (int d = 0; d < 4; d++) {
      int dh = d * 16 + col;
      Y[((size_t)(b * SEQ + qrow) * HEADS + h) * DH + dh] =
          f2bf(Oacc[d][r] * inv);
    }
  }
}

// ============================================================
// Kernel 3: out = Y @ w_o^T + b_o   (f32 out), async double-buffered
// ============================================================
__global__ __launch_bounds__(256) void k_oproj(
    const bf16* __restrict__ Yb, const bf16* __restrict__ wob,
    const float* __restrict__ bo, float* __restrict__ out)
{
  __shared__ bf16 sA[2][PM * SAS];
  __shared__ bf16 sB[2][PN * SBS];

  const int t    = threadIdx.x;
  const int lane = t & 31;
  const int wid  = t >> 5;
  const int hi   = lane >> 4;

  const int j0 = blockIdx.x * PN;
  const int m0 = blockIdx.y * PM;
  const int wm = (wid & 3) * 32;
  const int wn = (wid >> 2) * 32;

  v8f acc[2][2];
#pragma unroll
  for (int i = 0; i < 2; i++)
#pragma unroll
    for (int j = 0; j < 2; j++) acc[i][j] = zero8();

  auto issue = [&](int buf, int k0) {
#pragma unroll
    for (int it = 0; it < 2; it++) {
      int idx = t + it * 256;
      int row = idx >> 2, c = (idx & 3) * 8;
      async_b128(&sA[buf][row * SAS + c],
                 Yb + (size_t)(m0 + row) * EMB + k0 + c);
    }
    {
      int row = t >> 2, c = (t & 3) * 8;
      async_b128(&sB[buf][row * SBS + c],
                 wob + (size_t)(j0 + row) * EMB + k0 + c);
    }
  };

  issue(0, 0);
  int cur = 0;
  for (int k0 = 0; k0 < EMB; k0 += PKS) {
    if (k0 + PKS < EMB) { issue(cur ^ 1, k0 + PKS); WAIT_ASYNC("0x3"); }
    else                {                           WAIT_ASYNC("0x0"); }
    __syncthreads();
    v16bf a0 = load_a_frag(&sA[cur][(wm + 0)  * SAS], SAS);
    v16bf a1 = load_a_frag(&sA[cur][(wm + 16) * SAS], SAS);
    v16bf b0 = load_b_frag(&sB[cur][(wn + 0)  * SBS], SBS);
    v16bf b1 = load_b_frag(&sB[cur][(wn + 16) * SBS], SBS);
    acc[0][0] = wmma_bf16(a0, b0, acc[0][0]);
    acc[0][1] = wmma_bf16(a0, b1, acc[0][1]);
    acc[1][0] = wmma_bf16(a1, b0, acc[1][0]);
    acc[1][1] = wmma_bf16(a1, b1, acc[1][1]);
    __syncthreads();
    cur ^= 1;
  }

#pragma unroll
  for (int sm = 0; sm < 2; sm++)
#pragma unroll
    for (int sn = 0; sn < 2; sn++)
#pragma unroll
      for (int r = 0; r < 8; r++) {
        int gm = m0 + wm + sm * 16 + r + hi * 8;
        int gj = j0 + wn + sn * 16 + (lane & 15);
        out[(size_t)gm * EMB + gj] = acc[sm][sn][r] + bo[gj];
      }
}

// ============================================================
extern "C" void kernel_launch(void* const* d_in, const int* in_sizes, int n_in,
                              void* d_out, int out_size, void* d_ws, size_t ws_size,
                              hipStream_t stream) {
  (void)in_sizes; (void)n_in; (void)out_size; (void)ws_size;
  const float* z  = (const float*)d_in[0];
  const float* wq = (const float*)d_in[1];
  const float* wk = (const float*)d_in[2];
  const float* wv = (const float*)d_in[3];
  const float* wo = (const float*)d_in[4];
  const float* bo = (const float*)d_in[5];
  float* out = (float*)d_out;

  const size_t nZ = (size_t)MTOT * EMB;                 // 6,291,456
  const size_t nW = (size_t)EMB * EMB;                  //   589,824
  const size_t nQ = (size_t)BATCH * HEADS * SEQ * DH;   // 6,291,456

  bf16* zb  = (bf16*)d_ws;
  bf16* wqb = zb  + nZ;
  bf16* wkb = wqb + nW;
  bf16* wvb = wkb + nW;
  bf16* wob = wvb + nW;
  bf16* Q   = wob + nW;
  bf16* Kd  = Q   + nQ;
  bf16* Vt  = Kd  + nQ;
  bf16* Y   = Vt  + nQ;          // ~67.6 MB bf16 scratch total

  k_cvt<<<dim3((unsigned)(nZ / 2048)), 256, 0, stream>>>(z,  zb);
  k_cvt<<<dim3((unsigned)(nW / 2048)), 256, 0, stream>>>(wq, wqb);
  k_cvt<<<dim3((unsigned)(nW / 2048)), 256, 0, stream>>>(wk, wkb);
  k_cvt<<<dim3((unsigned)(nW / 2048)), 256, 0, stream>>>(wv, wvb);
  k_cvt<<<dim3((unsigned)(nW / 2048)), 256, 0, stream>>>(wo, wob);

  k_proj <<<dim3(EMB / PN, MTOT / PM, 3), 256, 0, stream>>>(zb, wqb, wkb, wvb,
                                                            Q, Kd, Vt);
  k_attn <<<dim3(SEQ / 128, BATCH * HEADS), 256, 0, stream>>>(Q, Kd, Vt, Y);
  k_oproj<<<dim3(EMB / PN, MTOT / PM),     256, 0, stream>>>(Y, wob, bo, out);
}
